// Struct2SeqLO_43946105372783
// MI455X (gfx1250) — matrix operand order repaired and checked by reading, not compiled, not run on hardware
//
#include <hip/hip_runtime.h>
#include <math.h>

// ---------------------------------------------------------------------------
// Struct2Seq forward for MI455X (gfx1250, wave32, WMMA).
// All dense GEMMs run on v_wmma_f32_16x16x32_bf16 (fp32 accumulate).
// Weights are pre-packed to bf16 in WMMA B-fragment order once per launch.
// The bandwidth-dominant K/V edge projections are fused: one wave per 16-row
// strip computes all 8 N-tiles for BOTH W_K and W_V from one A-fragment load,
// so the 189MB h_ESV tensor is read from HBM exactly once per layer.
// ---------------------------------------------------------------------------

typedef __bf16 bf16_t;
typedef __attribute__((ext_vector_type(16))) __bf16 v16bf;
typedef __attribute__((ext_vector_type(8)))  float  v8f;

#define BB 4
#define NN 1024
#define KK 30
#define HH 128
#define BN (BB*NN)          // 4096 nodes
#define BNK (BB*NN*KK)      // 122880 edges
#define VOCAB 20

// ---------------------------------------------------------------------------
// Pack fp32 weight (Kin x Nout, row-major) into bf16 WMMA B-fragment order:
//   layout [Kin/32][Nout/16][lane(32)][i(16)],
//   element(lane,i) = W[kt*32 + (i/8)*16 + (lane/16)*8 + i%8][nt*16 + lane%16]
// ---------------------------------------------------------------------------
__global__ __launch_bounds__(256)
void pack_w_kernel(const float* __restrict__ W, bf16_t* __restrict__ out,
                   int Kin, int Nout) {
    int gid = blockIdx.x * 256 + threadIdx.x;
    int total = Kin * Nout;
    if (gid >= total) return;
    int tiles_n = Nout >> 4;
    int per_kt  = tiles_n * 512;
    int kt  = gid / per_kt;
    int rem = gid - kt * per_kt;
    int nt  = rem >> 9;
    int e   = rem & 511;
    int lane = e >> 4, i = e & 15;
    int nn2 = lane & 15;
    int kk2 = ((i >> 3) << 4) + ((lane >> 4) << 3) + (i & 7);
    out[gid] = (bf16_t)W[(size_t)(kt * 32 + kk2) * Nout + nt * 16 + nn2];
}

// ---------------------------------------------------------------------------
// Generic WMMA GEMM: C[M,Nout] = act(A[M,Kin] * W + bias)
// One wave per 16x16 output tile; used for the small L2-resident GEMMs.
// A-fragment (ISA 7.12.2, 16-bit A 16x32): lane -> row m=lane%16,
//   element i -> k = (i/8)*16 + (lane/16)*8 + i%8  (two contiguous 8-runs).
// C/D: VGPR j -> m = j + 8*(lane/16), n = lane%16.
// ---------------------------------------------------------------------------
__global__ __launch_bounds__(256)
void wmma_gemm_kernel(const float* __restrict__ A, const bf16_t* __restrict__ Bp,
                      const float* __restrict__ bias, float* __restrict__ C,
                      int M, int Kin, int Nout, int relu) {
    const int lane = threadIdx.x & 31;
    const int wave = threadIdx.x >> 5;
    const int tiles_n = Nout >> 4;
    const int total   = (M >> 4) * tiles_n;
    const int tile = blockIdx.x * 8 + wave;
    if (tile >= total) return;                 // uniform per wave: EXEC stays all-ones
    const int mt = tile / tiles_n;
    const int nt = tile - mt * tiles_n;
    const int half = lane >> 4;
    const int mr   = lane & 15;

    const float*  arow = A + (size_t)(mt * 16 + mr) * Kin;
    const bf16_t* bcol = Bp + (size_t)nt * 512 + lane * 16;
    const size_t  bstride = (size_t)tiles_n * 512;

    v8f acc = {0.f, 0.f, 0.f, 0.f, 0.f, 0.f, 0.f, 0.f};
    for (int ko = 0; ko < Kin; ko += 32) {
        const float* p0 = arow + ko + half * 8;
        if (ko + 32 < Kin) __builtin_prefetch(p0 + 32, 0, 1);  // global_prefetch_b8
        v16bf a;
#pragma unroll
        for (int t = 0; t < 8; ++t) {
            a[t]     = (bf16_t)p0[t];        // k = ko + half*8 + t
            a[t + 8] = (bf16_t)p0[t + 16];   // k = ko + 16 + half*8 + t
        }
        v16bf b = *(const v16bf*)(bcol + (size_t)(ko >> 5) * bstride);
        acc = __builtin_amdgcn_wmma_f32_16x16x32_bf16(
                  false, a, false, b, (short)0, acc, false, false);
    }

    const int n = nt * 16 + mr;
    const float bv = bias ? bias[n] : 0.f;
#pragma unroll
    for (int j = 0; j < 8; ++j) {
        int m = mt * 16 + j + half * 8;
        float v = acc[j] + bv;
        if (relu) v = fmaxf(v, 0.f);
        C[(size_t)m * Nout + n] = v;
    }
}

// ---------------------------------------------------------------------------
// Fused K/V projection GEMM (Nout fixed = 128 = 8 tiles).
// One wave per 16-row strip of A; computes all 8 N-tiles for BOTH weights.
// 16 accumulators (128 VGPRs), 16 WMMAs per A-fragment load -> the huge edge
// tensor is streamed from HBM exactly once, weights stay L2-hot.
// ---------------------------------------------------------------------------
__global__ __launch_bounds__(256)
void wmma_gemm_kv_kernel(const float* __restrict__ A,
                         const bf16_t* __restrict__ BpK,
                         const bf16_t* __restrict__ BpV,
                         float* __restrict__ CK, float* __restrict__ CV,
                         int M, int Kin) {
    const int lane = threadIdx.x & 31;
    const int wave = threadIdx.x >> 5;
    const int strips = M >> 4;
    const int mt = blockIdx.x * 8 + wave;
    if (mt >= strips) return;                  // uniform per wave
    const int half = lane >> 4;
    const int mr   = lane & 15;
    const float* arow = A + (size_t)(mt * 16 + mr) * Kin;
    const size_t bstride = 8 * 512;            // tiles_n = 8

    v8f accK[8], accV[8];
#pragma unroll
    for (int i = 0; i < 8; ++i) {
        accK[i] = (v8f){0.f, 0.f, 0.f, 0.f, 0.f, 0.f, 0.f, 0.f};
        accV[i] = (v8f){0.f, 0.f, 0.f, 0.f, 0.f, 0.f, 0.f, 0.f};
    }
    for (int ko = 0; ko < Kin; ko += 32) {
        const float* p0 = arow + ko + half * 8;
        if (ko + 32 < Kin) __builtin_prefetch(p0 + 32, 0, 1);  // global_prefetch_b8
        v16bf a;
#pragma unroll
        for (int t = 0; t < 8; ++t) {
            a[t]     = (bf16_t)p0[t];
            a[t + 8] = (bf16_t)p0[t + 16];
        }
        const bf16_t* bk = BpK + (size_t)(ko >> 5) * bstride + lane * 16;
        const bf16_t* bv = BpV + (size_t)(ko >> 5) * bstride + lane * 16;
#pragma unroll
        for (int nt = 0; nt < 8; ++nt) {
            v16bf b1 = *(const v16bf*)(bk + nt * 512);
            accK[nt] = __builtin_amdgcn_wmma_f32_16x16x32_bf16(
                           false, a, false, b1, (short)0, accK[nt], false, false);
            v16bf b2 = *(const v16bf*)(bv + nt * 512);
            accV[nt] = __builtin_amdgcn_wmma_f32_16x16x32_bf16(
                           false, a, false, b2, (short)0, accV[nt], false, false);
        }
    }
#pragma unroll
    for (int nt = 0; nt < 8; ++nt) {
        const int n = nt * 16 + mr;
#pragma unroll
        for (int j = 0; j < 8; ++j) {
            int m = mt * 16 + j + half * 8;
            CK[(size_t)m * 128 + n] = accK[nt][j];
            CV[(size_t)m * 128 + n] = accV[nt][j];
        }
    }
}

// ---------------------------------------------------------------------------
// Encoder edge features: h_EV[r] = [ h_E[r] (128) | h_V[b, E_idx[r]] (128) ]
// ---------------------------------------------------------------------------
__global__ __launch_bounds__(256)
void build_hEV_kernel(const float* __restrict__ hE, const float* __restrict__ hV,
                      const int* __restrict__ Eidx, float* __restrict__ out) {
    size_t gid = (size_t)blockIdx.x * 256 + threadIdx.x;
    if (gid >= (size_t)BNK * 256) return;
    size_t r = gid >> 8;
    int c = (int)(gid & 255);
    float v;
    if (c < 128) {
        v = hE[r * 128 + c];
    } else {
        int b = (int)(r / (NN * KK));
        int j = Eidx[r];
        v = hV[((size_t)(b << 10) + j) * 128 + (c - 128)];
    }
    out[gid] = v;
}

// ---------------------------------------------------------------------------
// Decoder edge features with autoregressive masking:
//   h_ESV      = [h_E | gather(h_S) | gather(h_V_cur)]
//   h_ESV_enc  = [h_E | 0           | gather(h_V_enc)]
//   out = mask_bw * h_ESV + mask_fw * h_ESV_enc
//   ar = (rank[neighbor] < rank[node]); mask_bw = mask*ar; mask_fw = mask*(1-ar)
// ---------------------------------------------------------------------------
__global__ __launch_bounds__(256)
void build_hESV_kernel(const float* __restrict__ hE, const float* __restrict__ hS,
                       const float* __restrict__ hV, const float* __restrict__ hVenc,
                       const int* __restrict__ Eidx, const int* __restrict__ rank,
                       const float* __restrict__ mask, float* __restrict__ out) {
    size_t gid = (size_t)blockIdx.x * 256 + threadIdx.x;
    if (gid >= (size_t)BNK * 384) return;
    size_t r = gid / 384;
    int c = (int)(gid - r * 384);
    int b = (int)(r / (NN * KK));
    int n = (int)((r / KK) & (NN - 1));
    int node = (b << 10) + n;
    int j = Eidx[r];
    size_t nb = ((size_t)(b << 10) + j) * 128;
    float ar  = (rank[(b << 10) + j] < rank[node]) ? 1.f : 0.f;
    float mk  = mask[node];
    float mbw = mk * ar;
    float mfw = mk * (1.f - ar);
    float base, fw;
    if (c < 128)      { float he = hE[r * 128 + c]; base = he;            fw = mfw * he; }
    else if (c < 256) { base = hS[nb + (c - 128)];                        fw = 0.f; }
    else              { base = hV[nb + (c - 256)];  fw = mfw * hVenc[nb + (c - 256)]; }
    out[gid] = mbw * base + fw;
}

// ---------------------------------------------------------------------------
// Neighbor attention: one block (128 thr = 4 waves) per node; head == wave.
// logits = Q.Kt / sqrt(32); masked softmax over K=30; out = att . Vt
// ---------------------------------------------------------------------------
__global__ __launch_bounds__(128)
void attn_kernel(const float* __restrict__ Q, const float* __restrict__ Kt,
                 const float* __restrict__ Vt, const float* __restrict__ mask,
                 const int* __restrict__ Eidx, float* __restrict__ out,
                 int use_mask) {
    __shared__ float q[128];
    __shared__ float att[4][32];
    const int node = blockIdx.x;
    const int t = threadIdx.x;
    const int h = t >> 5, lane = t & 31;
    const size_t r0 = (size_t)node * KK;

    q[t] = Q[(size_t)node * 128 + t];
    __syncthreads();

    float logit = -3.402823466e38f;
    float mval  = 0.f;
    if (lane < KK) {
        const float* kr = Kt + (r0 + lane) * 128 + h * 32;
        float s = 0.f;
#pragma unroll
        for (int d = 0; d < 32; ++d) s += q[h * 32 + d] * kr[d];
        s *= 0.17677669529663687f;  // 1/sqrt(32)
        if (use_mask) {
            int b = node >> 10;
            mval = mask[node] * mask[(b << 10) + Eidx[r0 + lane]];
            logit = (mval > 0.f) ? s : -3.402823466e38f;
        } else {
            mval = 1.f; logit = s;
        }
    }
    float mx = logit;
    for (int off = 16; off; off >>= 1) mx = fmaxf(mx, __shfl_xor(mx, off, 32));
    float e = (lane < KK && logit > -1e37f) ? __expf(logit - mx) : 0.f;
    float sum = e;
    for (int off = 16; off; off >>= 1) sum += __shfl_xor(sum, off, 32);
    float a = (sum > 0.f) ? e / sum : 0.f;
    if (use_mask) a *= mval;
    att[h][lane] = a;
    __syncthreads();

    float o = 0.f;
    const float* vcol = Vt + r0 * 128 + t;
#pragma unroll
    for (int k = 0; k < KK; ++k) o += att[h][k] * vcol[(size_t)k * 128];
    out[(size_t)node * 128 + t] = o;
}

// ---------------------------------------------------------------------------
// Fused residual + LayerNorm (+ optional node mask), in-place on hV.
// ---------------------------------------------------------------------------
__global__ __launch_bounds__(128)
void ln_residual_kernel(float* __restrict__ hV, const float* __restrict__ dh,
                        const float* __restrict__ g, const float* __restrict__ bta,
                        const float* __restrict__ mask, int apply_mask) {
    __shared__ float red[4];
    const int row = blockIdx.x, t = threadIdx.x;
    float x = hV[(size_t)row * 128 + t] + dh[(size_t)row * 128 + t];
    float s = x;
    for (int off = 16; off; off >>= 1) s += __shfl_xor(s, off, 32);
    if ((t & 31) == 0) red[t >> 5] = s;
    __syncthreads();
    float mu = (red[0] + red[1] + red[2] + red[3]) * (1.f / 128.f);
    __syncthreads();
    float d = x - mu;
    float v = d * d;
    for (int off = 16; off; off >>= 1) v += __shfl_xor(v, off, 32);
    if ((t & 31) == 0) red[t >> 5] = v;
    __syncthreads();
    float var = (red[0] + red[1] + red[2] + red[3]) * (1.f / 128.f);
    float y = d * __frsqrt_rn(var + 1e-6f) * g[t] + bta[t];
    if (apply_mask) y *= mask[row];
    hV[(size_t)row * 128 + t] = y;
}

// rank = argsort(permutation) = inverse permutation
__global__ __launch_bounds__(256)
void rank_kernel(const int* __restrict__ perm, int* __restrict__ rank) {
    int t = blockIdx.x * 256 + threadIdx.x;
    if (t >= BN) return;
    int b = t >> 10, i = t & 1023;
    rank[(b << 10) + perm[t]] = i;
}

// h_S = W_s[S]
__global__ __launch_bounds__(256)
void embed_kernel(const int* __restrict__ S, const float* __restrict__ Ws,
                  float* __restrict__ hS) {
    int gid = blockIdx.x * 256 + threadIdx.x;
    if (gid >= BN * 128) return;
    int node = gid >> 7, c = gid & 127;
    hS[gid] = Ws[S[node] * 128 + c];
}

// logits = hV @ W_out + b_out ; log_softmax over VOCAB=20. One wave per node.
__global__ __launch_bounds__(32)
void logits_kernel(const float* __restrict__ hV, const float* __restrict__ Wout,
                   const float* __restrict__ bout, float* __restrict__ outLP) {
    const int node = blockIdx.x, lane = threadIdx.x;
    float l = -3.402823466e38f;
    if (lane < VOCAB) {
        float s = bout[lane];
        const float* hv = hV + (size_t)node * 128;
        for (int k = 0; k < 128; ++k) s += hv[k] * Wout[k * VOCAB + lane];
        l = s;
    }
    float mx = l;
    for (int off = 16; off; off >>= 1) mx = fmaxf(mx, __shfl_xor(mx, off, 32));
    float e = (lane < VOCAB) ? __expf(l - mx) : 0.f;
    float sum = e;
    for (int off = 16; off; off >>= 1) sum += __shfl_xor(sum, off, 32);
    float lse = mx + __logf(sum);
    if (lane < VOCAB) outLP[(size_t)node * VOCAB + lane] = l - lse;
}

// p_order = h_p @ Wp2 + bp2, masked with -inf
__global__ __launch_bounds__(128)
void porder_kernel(const float* __restrict__ hp, const float* __restrict__ Wp2,
                   const float* __restrict__ bp2, const float* __restrict__ mask,
                   float* __restrict__ out) {
    __shared__ float red[4];
    const int node = blockIdx.x, t = threadIdx.x;
    float s = hp[(size_t)node * 128 + t] * Wp2[t];
    for (int off = 16; off; off >>= 1) s += __shfl_xor(s, off, 32);
    if ((t & 31) == 0) red[t >> 5] = s;
    __syncthreads();
    if (t == 0) {
        float v = red[0] + red[1] + red[2] + red[3] + bp2[0];
        out[node] = (mask[node] > 0.f) ? v : -INFINITY;
    }
}

// ---------------------------------------------------------------------------
// Host orchestration.
// Input leaf order assumption: setup_inputs() dict insertion order, with the
// nested `params` pytree flattened in its own insertion order:
//   0:V 1:E 2:mask 3:W_v 4:b_v 5:W_e 6:b_e 7:W_s 8:W_out 9:b_out 10:Wp1
//   11:bp1 12:Wp2 13:bp2 14..49: enc[0..2] x {W_Q,W_K,W_V,W_O,ff_W1,ff_b1,
//   ff_W2,ff_b2,n1_g,n1_b,n2_g,n2_b}, 50..85: dec likewise, 86:S 87:E_idx
//   88:permutation.  (n_in == 89)
// Workspace requirement: ~420 MB (dominated by the 384-col h_ESV buffer).
// ---------------------------------------------------------------------------
struct LayerPk {
    const bf16_t *WQ, *WK, *WV, *WO, *W1, *W2;
    const float *b1, *b2, *g1, *bt1, *g2, *bt2;
};

extern "C" void kernel_launch(void* const* d_in, const int* in_sizes, int n_in,
                              void* d_out, int out_size, void* d_ws, size_t ws_size,
                              hipStream_t stream) {
    const float* V     = (const float*)d_in[0];
    const float* E     = (const float*)d_in[1];
    const float* mask  = (const float*)d_in[2];
    const float* W_v   = (const float*)d_in[3];
    const float* b_v   = (const float*)d_in[4];
    const float* W_e   = (const float*)d_in[5];
    const float* b_e   = (const float*)d_in[6];
    const float* W_s   = (const float*)d_in[7];
    const float* W_out = (const float*)d_in[8];
    const float* b_out = (const float*)d_in[9];
    const float* Wp1   = (const float*)d_in[10];
    const float* bp1   = (const float*)d_in[11];
    const float* Wp2   = (const float*)d_in[12];
    const float* bp2   = (const float*)d_in[13];
    const int ENC0 = 14, DEC0 = 50;
    const int* S    = (const int*)d_in[86];
    const int* Eix  = (const int*)d_in[87];
    const int* perm = (const int*)d_in[88];

    // ---- workspace bump allocator (256B aligned) ----
    char* wsp = (char*)d_ws;
    size_t off = 0;
    auto alloc = [&](size_t bytes) -> void* {
        void* p = wsp + off;
        off = (off + bytes + 255) & ~(size_t)255;
        return p;
    };

    float* hV    = (float*)alloc((size_t)BN * 128 * 4);
    float* hVenc = (float*)alloc((size_t)BN * 128 * 4);
    float* hS    = (float*)alloc((size_t)BN * 128 * 4);
    float* Qb    = (float*)alloc((size_t)BN * 128 * 4);
    float* attn  = (float*)alloc((size_t)BN * 128 * 4);
    float* dh    = (float*)alloc((size_t)BN * 128 * 4);
    float* hp    = (float*)alloc((size_t)BN * 128 * 4);
    float* ff1   = (float*)alloc((size_t)BN * 512 * 4);
    float* hE    = (float*)alloc((size_t)BNK * 128 * 4);
    float* Ktb   = (float*)alloc((size_t)BNK * 128 * 4);
    float* Vtb   = (float*)alloc((size_t)BNK * 128 * 4);
    float* hEV   = (float*)alloc((size_t)BNK * 384 * 4);
    int*   rank  = (int*)alloc((size_t)BN * 4);

    auto pack = [&](const float* W, int Kin, int Nout) -> const bf16_t* {
        bf16_t* dst = (bf16_t*)alloc((size_t)Kin * Nout * 2);
        int total = Kin * Nout;
        pack_w_kernel<<<(total + 255) / 256, 256, 0, stream>>>(W, dst, Kin, Nout);
        return dst;
    };
    auto gemm = [&](const float* A, const bf16_t* Bp, const float* bias, float* C,
                    int M, int Kin, int Nout, int relu) {
        int tiles = (M >> 4) * (Nout >> 4);
        wmma_gemm_kernel<<<(tiles + 7) / 8, 256, 0, stream>>>(A, Bp, bias, C,
                                                              M, Kin, Nout, relu);
    };

    // ---- pack all GEMM weights to bf16 fragments ----
    const bf16_t* Wv_p  = pack(W_v, 128, 128);
    const bf16_t* We_p  = pack(W_e, 128, 128);
    const bf16_t* Wp1_p = pack(Wp1, 128, 128);
    LayerPk L[6];
    for (int i = 0; i < 6; ++i) {
        int base = (i < 3) ? (ENC0 + 12 * i) : (DEC0 + 12 * (i - 3));
        int kin_kv = (i < 3) ? 256 : 384;
        L[i].WQ  = pack((const float*)d_in[base + 0], 128, 128);
        L[i].WK  = pack((const float*)d_in[base + 1], kin_kv, 128);
        L[i].WV  = pack((const float*)d_in[base + 2], kin_kv, 128);
        L[i].WO  = pack((const float*)d_in[base + 3], 128, 128);
        L[i].W1  = pack((const float*)d_in[base + 4], 128, 512);
        L[i].b1  = (const float*)d_in[base + 5];
        L[i].W2  = pack((const float*)d_in[base + 6], 512, 128);
        L[i].b2  = (const float*)d_in[base + 7];
        L[i].g1  = (const float*)d_in[base + 8];
        L[i].bt1 = (const float*)d_in[base + 9];
        L[i].g2  = (const float*)d_in[base + 10];
        L[i].bt2 = (const float*)d_in[base + 11];
    }

    // ---- input projections ----
    gemm(V, Wv_p, b_v, hV, BN, 128, 128, 0);
    gemm(E, We_p, b_e, hE, BNK, 128, 128, 0);

    auto run_layer = [&](int li, int kin_kv, int use_mask) {
        gemm(hV, L[li].WQ, nullptr, Qb, BN, 128, 128, 0);
        // fused K+V projection of the edge tensor: A streamed once for both
        wmma_gemm_kv_kernel<<<(BNK / 16 + 7) / 8, 256, 0, stream>>>(
            hEV, L[li].WK, L[li].WV, Ktb, Vtb, BNK, kin_kv);
        attn_kernel<<<BN, 128, 0, stream>>>(Qb, Ktb, Vtb, mask, Eix, attn, use_mask);
        gemm(attn, L[li].WO, nullptr, dh, BN, 128, 128, 0);
        ln_residual_kernel<<<BN, 128, 0, stream>>>(hV, dh, L[li].g1, L[li].bt1, mask, 0);
        gemm(hV, L[li].W1, L[li].b1, ff1, BN, 128, 512, 1);
        gemm(ff1, L[li].W2, L[li].b2, dh, BN, 512, 128, 0);
        ln_residual_kernel<<<BN, 128, 0, stream>>>(hV, dh, L[li].g2, L[li].bt2, mask, 1);
    };

    // ---- encoder ----
    for (int li = 0; li < 3; ++li) {
        size_t tot = (size_t)BNK * 256;
        build_hEV_kernel<<<(unsigned)((tot + 255) / 256), 256, 0, stream>>>(hE, hV, Eix, hEV);
        run_layer(li, 256, /*use_mask=*/1);
    }
    hipMemcpyAsync(hVenc, hV, (size_t)BN * 128 * 4, hipMemcpyDeviceToDevice, stream);

    // ---- decoder prep ----
    rank_kernel<<<(BN + 255) / 256, 256, 0, stream>>>(perm, rank);
    embed_kernel<<<(BN * 128 + 255) / 256, 256, 0, stream>>>(S, W_s, hS);

    // ---- decoder ----
    for (int li = 3; li < 6; ++li) {
        size_t tot = (size_t)BNK * 384;
        build_hESV_kernel<<<(unsigned)((tot + 255) / 256), 256, 0, stream>>>(
            hE, hS, hV, hVenc, Eix, rank, mask, hEV);
        run_layer(li, 384, /*use_mask=*/0);
    }

    // ---- output heads ----
    float* outp = (float*)d_out;
    logits_kernel<<<BN, 32, 0, stream>>>(hV, W_out, b_out, outp);
    gemm(hV, Wp1_p, bp1, hp, BN, 128, 128, 1);
    porder_kernel<<<BN, 128, 0, stream>>>(hp, Wp2, bp2, mask, outp + (size_t)BN * VOCAB);
}
